// MultiHeadAttention_22978075034351
// MI455X (gfx1250) — matrix operand order repaired and checked by reading, not compiled, not run on hardware
//
#include <hip/hip_runtime.h>

// ---------------------------------------------------------------------------
// MHA forward for MI455X (gfx1250, wave32, WMMA).
//   D_MODEL=1024, NUM_HEADS=16, D_K=64, B=2, S=2048, causal + pad mask.
// Pipeline: f32->bf16 convert, 3 WMMA projection GEMMs, flash-attention
// (online softmax) WMMA kernel with async-to-LDS double-buffered K/V
// staging, final WMMA output projection (f32 out).
// All matmuls arranged as A @ W^T so both WMMA operands load contiguous
// row chunks (B-fragment columns == rows of W / K / V^T). No transposes.
// Workspace use: ~64 MiB of d_ws.
// ---------------------------------------------------------------------------

#define SEQ   2048
#define DM    1024
#define NH    16
#define DK    64
#define BATCH 2
#define MTOK  (BATCH * SEQ)   // 4096 token rows

typedef unsigned short u16;
typedef __attribute__((ext_vector_type(16))) __bf16 v16bf;
typedef __attribute__((ext_vector_type(8)))  float  v8f;

struct U32x8 { uint4 lo, hi; };

static __device__ __forceinline__ v16bf make_frag(uint4 lo, uint4 hi) {
  U32x8 t{lo, hi};
  return __builtin_bit_cast(v16bf, t);
}

// f32 -> bf16, round-to-nearest-even
static __device__ __forceinline__ u16 f2bf(float f) {
  unsigned u = __builtin_bit_cast(unsigned, f);
  u += 0x7FFFu + ((u >> 16) & 1u);
  return (u16)(u >> 16);
}

static __device__ __forceinline__ v8f vzero8() {
  v8f z = {0.f, 0.f, 0.f, 0.f, 0.f, 0.f, 0.f, 0.f};
  return z;
}

// Async global->LDS copy of 16 bytes (tracked by ASYNCcnt).
static __device__ __forceinline__ void async_cp_b128(const u16* gsrc, u16* lds_dst) {
  const unsigned lds = (unsigned)(uintptr_t)lds_dst;     // low 32b of generic = LDS addr
  const unsigned long long ga = (unsigned long long)(uintptr_t)gsrc;
  asm volatile("global_load_async_to_lds_b128 %0, %1, off"
               :: "v"(lds), "v"(ga) : "memory");
}

// ---------------------------------------------------------------------------
// f32 -> bf16 bulk convert (vectorized float4 -> ushort4)
// ---------------------------------------------------------------------------
__global__ __launch_bounds__(256) void mha_cvt_bf16(const float* __restrict__ src,
                                                    u16* __restrict__ dst, int n4) {
  int i = blockIdx.x * blockDim.x + threadIdx.x;
  if (i < n4) {
    float4 v = reinterpret_cast<const float4*>(src)[i];
    ushort4 o;
    o.x = f2bf(v.x); o.y = f2bf(v.y); o.z = f2bf(v.z); o.w = f2bf(v.w);
    reinterpret_cast<ushort4*>(dst)[i] = o;
  }
}

// ---------------------------------------------------------------------------
// GEMM: C[m,n] = sum_k X[m,k] * W[n,k]   (i.e. X @ W^T)
//   X: [4096,1024] bf16 row-major, W: [1024,1024] bf16 row-major.
// Wave tile 64x32 (4x2 wmma tiles), 8 waves/WG -> WG tile 256x64.
// Weights are fully L2-resident (2 MB bf16 vs 192 MB L2) -> direct b128 loads.
// MODE 0: write bf16 head-split [B,H,S,DK]            (Q, K projections)
// MODE 1: write bf16 head-split transposed [B,H,DK,S] (V projection)
// MODE 2: write f32 row-major [4096,1024]             (final output proj)
// ---------------------------------------------------------------------------
template <int MODE>
__global__ __launch_bounds__(256) void mha_gemm_bf16(const u16* __restrict__ X,
                                                     const u16* __restrict__ W,
                                                     void* __restrict__ out) {
  const int lane = threadIdx.x & 31;
  const int wave = threadIdx.x >> 5;
  const int half = lane >> 4;   // lane-half: selects K sub-range of fragments
  const int l16  = lane & 15;

  const int m_base = blockIdx.x * 256 + (wave & 3) * 64;
  const int n_base = blockIdx.y * 64  + (wave >> 2) * 32;

  v8f acc[4][2];
#pragma unroll
  for (int mt = 0; mt < 4; ++mt)
#pragma unroll
    for (int nt = 0; nt < 2; ++nt) acc[mt][nt] = vzero8();

  for (int kk = 0; kk < DM; kk += 32) {
    // A fragments: lane holds row m, K = {base..base+7, base+16..base+23}, base=half*8
    v16bf afrag[4];
#pragma unroll
    for (int mt = 0; mt < 4; ++mt) {
      const u16* row = X + (size_t)(m_base + mt * 16 + l16) * DM + kk + half * 8;
      uint4 lo = *reinterpret_cast<const uint4*>(row);
      uint4 hi = *reinterpret_cast<const uint4*>(row + 16);
      afrag[mt] = make_frag(lo, hi);
    }
    // B fragments: lane holds column n (= row n of W), K = {base..base+7, base+16..base+23}, base=half*8
    v16bf bfrag[2];
#pragma unroll
    for (int nt = 0; nt < 2; ++nt) {
      const u16* row = W + (size_t)(n_base + nt * 16 + l16) * DM + kk + half * 8;
      uint4 lo = *reinterpret_cast<const uint4*>(row);
      uint4 hi = *reinterpret_cast<const uint4*>(row + 16);
      bfrag[nt] = make_frag(lo, hi);
    }
    if (kk + 32 < DM) {   // warm caches for next K-step (global_prefetch_b8)
      __builtin_prefetch(X + (size_t)(m_base + l16) * DM + kk + 32, 0, 1);
      __builtin_prefetch(W + (size_t)(n_base + l16) * DM + kk + 32, 0, 1);
    }
#pragma unroll
    for (int mt = 0; mt < 4; ++mt)
#pragma unroll
      for (int nt = 0; nt < 2; ++nt)
        acc[mt][nt] = __builtin_amdgcn_wmma_f32_16x16x32_bf16(
            false, afrag[mt], false, bfrag[nt], (short)0, acc[mt][nt], false, false);
  }

  // Epilogue. C layout: lane holds column n_local=l16; rows r + 8*half per VGPR r.
#pragma unroll
  for (int mt = 0; mt < 4; ++mt) {
#pragma unroll
    for (int nt = 0; nt < 2; ++nt) {
#pragma unroll
      for (int r = 0; r < 8; ++r) {
        const int m = m_base + mt * 16 + r + half * 8;
        const int n = n_base + nt * 16 + l16;
        const float val = acc[mt][nt][r];
        if constexpr (MODE == 0) {
          const int b = m >> 11, s = m & (SEQ - 1);
          const int h = n >> 6, d = n & (DK - 1);
          ((u16*)out)[(((size_t)(b * NH + h) * SEQ) + s) * DK + d] = f2bf(val);
        } else if constexpr (MODE == 1) {
          const int b = m >> 11, s = m & (SEQ - 1);
          const int h = n >> 6, d = n & (DK - 1);
          ((u16*)out)[(((size_t)(b * NH + h) * DK) + d) * SEQ + s] = f2bf(val);
        } else {
          ((float*)out)[(size_t)m * DM + n] = val;
        }
      }
    }
  }
}

// ---------------------------------------------------------------------------
// Flash attention. One WG = 256 threads = 8 waves, all on one (b,h); wave w
// owns query rows [q0, q0+32). K/V 32-key chunks are staged ONCE per WG into
// double-buffered LDS with global_load_async_to_lds_b128 (ASYNCcnt pipeline),
// then consumed as WMMA fragments via ds_load_b128. Online softmax keeps row
// stats per-lane in the WMMA C layout; causal compare only on the diagonal
// chunk; waves past their causal horizon idle at the barriers.
// ---------------------------------------------------------------------------
__global__ __launch_bounds__(256) void mha_flash_attn(const u16* __restrict__ Qh,
                                                      const u16* __restrict__ Kh,
                                                      const u16* __restrict__ Vt,
                                                      const int* __restrict__ mask,
                                                      u16* __restrict__ attn) {
  __shared__ __align__(16) u16 Kbuf[2][32 * DK];   // 2 x 4 KB, [row][d]
  __shared__ __align__(16) u16 Vbuf[2][DK * 32];   // 2 x 4 KB, [d][key]
  __shared__ __align__(16) u16 pbuf[8][32 * 32];   // 2 KB per wave (P restage)

  const int tid  = threadIdx.x;
  const int lane = tid & 31;
  const int wave = tid >> 5;
  const int half = lane >> 4;
  const int l16  = lane & 15;

  const int b  = blockIdx.z;
  const int h  = blockIdx.y;
  const int q0 = blockIdx.x * 256 + wave * 32;

  const u16* Qp = Qh + (size_t)(b * NH + h) * SEQ * DK;
  const u16* Kp = Kh + (size_t)(b * NH + h) * SEQ * DK;
  const u16* Vp = Vt + (size_t)(b * NH + h) * DK * SEQ;
  const int* mrow = mask + b * SEQ;

  // WG-wide async staging of chunk j into buffer `buf`: 2 b128 per thread.
  const int krow = tid >> 3, kseg = tid & 7;   // K: 32 rows x 128 B
  const int vrow = tid >> 2, vseg = tid & 3;   // V: 64 rows x 64 B
  auto stage = [&](int j, int buf) {
    async_cp_b128(Kp + (size_t)(j * 32 + krow) * DK + kseg * 8,
                  &Kbuf[buf][krow * DK + kseg * 8]);
    async_cp_b128(Vp + (size_t)vrow * SEQ + j * 32 + vseg * 8,
                  &Vbuf[buf][vrow * 32 + vseg * 8]);
  };

  // Q fragments, loaded once: [mtile][kstep over DK=64]
  v16bf qf[2][2];
#pragma unroll
  for (int mt = 0; mt < 2; ++mt)
#pragma unroll
    for (int ks = 0; ks < 2; ++ks) {
      const u16* row = Qp + (size_t)(q0 + mt * 16 + l16) * DK + ks * 32 + half * 8;
      qf[mt][ks] = make_frag(*reinterpret_cast<const uint4*>(row),
                             *reinterpret_cast<const uint4*>(row + 16));
    }

  v8f oacc[2][4];
#pragma unroll
  for (int mt = 0; mt < 2; ++mt)
#pragma unroll
    for (int nt = 0; nt < 4; ++nt) oacc[mt][nt] = vzero8();

  float mi[2][8], li[2][8];
#pragma unroll
  for (int mt = 0; mt < 2; ++mt)
#pragma unroll
    for (int r = 0; r < 8; ++r) { mi[mt][r] = -1e30f; li[mt][r] = 0.f; }

  const int jmax_me = q0 >> 5;                 // my causal horizon (diag chunk)
  const int jmax_wg = blockIdx.x * 8 + 7;      // WG-uniform trip count

  stage(0, 0);
  for (int j = 0; j <= jmax_wg; ++j) {
    const int cur = j & 1;
    if (j < jmax_wg) {
      stage(j + 1, cur ^ 1);
      asm volatile("s_wait_asynccnt 2" ::: "memory");   // chunk j landed
    } else {
      asm volatile("s_wait_asynccnt 0" ::: "memory");
    }
    __syncthreads();

    if (j <= jmax_me) {
      const int k0 = j * 32;

      // ---- scores: S = Q @ K_chunk^T (32 queries x 32 keys), K from LDS ----
      v16bf kf[2][2];
#pragma unroll
      for (int nt = 0; nt < 2; ++nt)
#pragma unroll
        for (int ks = 0; ks < 2; ++ks) {
          const u16* row = &Kbuf[cur][(nt * 16 + l16) * DK + ks * 32 + half * 8];
          kf[nt][ks] = make_frag(*reinterpret_cast<const uint4*>(row),
                                 *reinterpret_cast<const uint4*>(row + 16));
        }
      v8f sc[2][2];
#pragma unroll
      for (int mt = 0; mt < 2; ++mt)
#pragma unroll
        for (int nt = 0; nt < 2; ++nt) {
          v8f s = vzero8();
          s = __builtin_amdgcn_wmma_f32_16x16x32_bf16(false, qf[mt][0], false, kf[nt][0],
                                                      (short)0, s, false, false);
          s = __builtin_amdgcn_wmma_f32_16x16x32_bf16(false, qf[mt][1], false, kf[nt][1],
                                                      (short)0, s, false, false);
          sc[mt][nt] = s;
        }

      // ---- mask + online softmax; P written to per-wave LDS as bf16 ----
      const int n0 = k0 + l16;
      const int n1 = n0 + 16;
      const bool diag = (j == jmax_me);        // only chunk needing causal cmp
      const bool dead0 = (mrow[n0] == 0);
      const bool dead1 = (mrow[n1] == 0);
#pragma unroll
      for (int mt = 0; mt < 2; ++mt) {
#pragma unroll
        for (int r = 0; r < 8; ++r) {
          float s0 = sc[mt][0][r] * 0.125f;
          float s1 = sc[mt][1][r] * 0.125f;
          if (dead0) s0 = -10000.f;
          if (dead1) s1 = -10000.f;
          if (diag) {
            const int q = q0 + mt * 16 + r + half * 8;
            if (n0 > q) s0 = -10000.f;
            if (n1 > q) s1 = -10000.f;
          }
          // row max across the 16 lanes of this half (butterfly)
          float cm = fmaxf(s0, s1);
          cm = fmaxf(cm, __shfl_xor(cm, 1));
          cm = fmaxf(cm, __shfl_xor(cm, 2));
          cm = fmaxf(cm, __shfl_xor(cm, 4));
          cm = fmaxf(cm, __shfl_xor(cm, 8));
          const float nm    = fmaxf(mi[mt][r], cm);
          const float alpha = __expf(mi[mt][r] - nm);
          const float p0 = __expf(s0 - nm);
          const float p1 = __expf(s1 - nm);
          float rs = p0 + p1;
          rs += __shfl_xor(rs, 1);
          rs += __shfl_xor(rs, 2);
          rs += __shfl_xor(rs, 4);
          rs += __shfl_xor(rs, 8);
          li[mt][r] = li[mt][r] * alpha + rs;
          mi[mt][r] = nm;
#pragma unroll
          for (int nt = 0; nt < 4; ++nt) oacc[mt][nt][r] *= alpha;
          const int prow = mt * 16 + r + half * 8;
          pbuf[wave][prow * 32 + l16]      = f2bf(p0);
          pbuf[wave][prow * 32 + 16 + l16] = f2bf(p1);
        }
      }

      // ---- out += P @ V_chunk (K-dim = 32 keys), V from LDS ----
      v16bf vf[4];
#pragma unroll
      for (int nt = 0; nt < 4; ++nt) {
        const u16* row = &Vbuf[cur][(nt * 16 + l16) * 32 + half * 8];
        vf[nt] = make_frag(*reinterpret_cast<const uint4*>(row),
                           *reinterpret_cast<const uint4*>(row + 16));
      }
#pragma unroll
      for (int mt = 0; mt < 2; ++mt) {
        const u16* pr = &pbuf[wave][(mt * 16 + l16) * 32 + half * 8];
        v16bf pf = make_frag(*reinterpret_cast<const uint4*>(pr),
                             *reinterpret_cast<const uint4*>(pr + 16));
#pragma unroll
        for (int nt = 0; nt < 4; ++nt)
          oacc[mt][nt] = __builtin_amdgcn_wmma_f32_16x16x32_bf16(
              false, pf, false, vf[nt], (short)0, oacc[mt][nt], false, false);
      }
    }
    __syncthreads();   // protect buffers before next-step async overwrite
  }

  // ---- normalize and store to [B, S, H*DK] bf16 ----
#pragma unroll
  for (int mt = 0; mt < 2; ++mt) {
#pragma unroll
    for (int nt = 0; nt < 4; ++nt) {
#pragma unroll
      for (int r = 0; r < 8; ++r) {
        const int q = q0 + mt * 16 + r + half * 8;
        const int d = nt * 16 + l16;
        const float val = oacc[mt][nt][r] / li[mt][r];
        attn[((size_t)(b * SEQ) + q) * DM + h * DK + d] = f2bf(val);
      }
    }
  }
}

// ---------------------------------------------------------------------------
// Launch
// ---------------------------------------------------------------------------
extern "C" void kernel_launch(void* const* d_in, const int* in_sizes, int n_in,
                              void* d_out, int out_size, void* d_ws, size_t ws_size,
                              hipStream_t stream) {
  const float* query = (const float*)d_in[0];
  const float* key   = (const float*)d_in[1];
  const float* value = (const float*)d_in[2];
  const int*   maskp = (const int*)  d_in[3];
  const float* Wq    = (const float*)d_in[4];
  const float* Wk    = (const float*)d_in[5];
  const float* Wv    = (const float*)d_in[6];
  const float* Wo    = (const float*)d_in[7];

  // carve ~64 MiB of workspace
  char* p = (char*)d_ws;
  auto take = [&](size_t elems) {
    u16* r = (u16*)p;
    p += (elems * sizeof(u16) + 255) & ~(size_t)255;
    return r;
  };
  u16* bq   = take((size_t)MTOK * DM);
  u16* bk   = take((size_t)MTOK * DM);
  u16* bv   = take((size_t)MTOK * DM);
  u16* bWq  = take((size_t)DM * DM);
  u16* bWk  = take((size_t)DM * DM);
  u16* bWv  = take((size_t)DM * DM);
  u16* bWo  = take((size_t)DM * DM);
  u16* Qh   = take((size_t)MTOK * DM);   // [B,H,S,DK]
  u16* Kh   = take((size_t)MTOK * DM);   // [B,H,S,DK]
  u16* Vtw  = take((size_t)MTOK * DM);   // [B,H,DK,S]
  u16* attn = take((size_t)MTOK * DM);   // [B,S,DM]

  const dim3 blk(256);
  const int nAct = MTOK * DM / 4, nW = DM * DM / 4;
  mha_cvt_bf16<<<dim3((nAct + 255) / 256), blk, 0, stream>>>(query, bq, nAct);
  mha_cvt_bf16<<<dim3((nAct + 255) / 256), blk, 0, stream>>>(key,   bk, nAct);
  mha_cvt_bf16<<<dim3((nAct + 255) / 256), blk, 0, stream>>>(value, bv, nAct);
  mha_cvt_bf16<<<dim3((nW + 255) / 256),   blk, 0, stream>>>(Wq, bWq, nW);
  mha_cvt_bf16<<<dim3((nW + 255) / 256),   blk, 0, stream>>>(Wk, bWk, nW);
  mha_cvt_bf16<<<dim3((nW + 255) / 256),   blk, 0, stream>>>(Wv, bWv, nW);
  mha_cvt_bf16<<<dim3((nW + 255) / 256),   blk, 0, stream>>>(Wo, bWo, nW);

  const dim3 ggrid(MTOK / 256, DM / 64);   // 16 x 16
  mha_gemm_bf16<0><<<ggrid, blk, 0, stream>>>(bq, bWq, Qh);
  mha_gemm_bf16<0><<<ggrid, blk, 0, stream>>>(bk, bWk, Kh);
  mha_gemm_bf16<1><<<ggrid, blk, 0, stream>>>(bv, bWv, Vtw);

  mha_flash_attn<<<dim3(SEQ / 256, NH, BATCH), blk, 0, stream>>>(Qh, Kh, Vtw, maskp, attn);

  mha_gemm_bf16<2><<<ggrid, blk, 0, stream>>>(attn, bWo, d_out);
}
